// GE2ELoss_64226940944543
// MI455X (gfx1250) — matrix operand order repaired and checked by reading, not compiled, not run on hardware
//
#include <hip/hip_runtime.h>
#include <hip/hip_bf16.h>

// ---------------------------------------------------------------------------
// GE2E loss on gfx1250 (MI455X): centroids via L2 atomics, logits GEMM via
// bf16 hi/lo split-precision V_WMMA_F32_16X16X32_BF16 (3 wmma per k-step for
// near-f32 accuracy), fused register-resident log-softmax + NLL reduction.
// ---------------------------------------------------------------------------

#define DDIM 256
#define CNUM 1024

typedef __attribute__((ext_vector_type(16))) __bf16 v16bf;
typedef __attribute__((ext_vector_type(8)))  float  v8f;
typedef __attribute__((ext_vector_type(8)))  unsigned short v8u16;

union AFrag { v16bf v; unsigned short s[16]; };
union BFrag { v16bf v; v8u16 h[2]; };

static __device__ inline v8f wmma_bf16(v16bf a, v16bf b, v8f c) {
  return __builtin_amdgcn_wmma_f32_16x16x32_bf16(
      /*neg_a=*/false, a, /*neg_b=*/false, b,
      /*c_mod=*/(short)0, c, /*reuse_a=*/false, /*reuse_b=*/false);
}

// Round-to-nearest-even f32 -> bf16 split: x ~= hi + lo (both bf16).
static __device__ inline void bf16_split(float x, unsigned short& hi,
                                         unsigned short& lo) {
  union { float f; unsigned u; } a; a.f = x;
  unsigned r = a.u + 0x7FFFu + ((a.u >> 16) & 1u);
  hi = (unsigned short)(r >> 16);
  union { unsigned u; float f; } hf; hf.u = ((unsigned)hi) << 16;
  float rem = x - hf.f;
  union { float f; unsigned u; } b2; b2.f = rem;
  unsigned r2 = b2.u + 0x7FFFu + ((b2.u >> 16) & 1u);
  lo = (unsigned short)(r2 >> 16);
}

static __device__ inline void split4(float4 c, unsigned short* hs,
                                     unsigned short* ls) {
  bf16_split(c.x, hs[0], ls[0]);
  bf16_split(c.y, hs[1], ls[1]);
  bf16_split(c.z, hs[2], ls[2]);
  bf16_split(c.w, hs[3], ls[3]);
}

// ---------------------------------------------------------------------------
// Kernel 1: zero the accumulation region of the workspace.
// ---------------------------------------------------------------------------
__global__ void ge2e_zero(float* p, int nfloats) {
  int i = blockIdx.x * blockDim.x + threadIdx.x;
  if (i < nfloats) p[i] = 0.0f;
}

// ---------------------------------------------------------------------------
// Kernel 2: segment sums + counts. One block (=256 threads = D) per row, so
// the label load is wave-uniform; f32 atomics land in the 1MB L2-resident
// sums buffer.
// ---------------------------------------------------------------------------
__global__ __launch_bounds__(256) void ge2e_segsum(
    const float* __restrict__ emb, const int* __restrict__ labels,
    float* __restrict__ sums, float* __restrict__ counts) {
  int n = blockIdx.x;
  int d = threadIdx.x;
  int lab = labels[n];
  atomicAdd(&sums[lab * DDIM + d], emb[(size_t)n * DDIM + d]);
  if (d == 0) atomicAdd(&counts[lab], 1.0f);
}

// ---------------------------------------------------------------------------
// Kernel 3: centroids = sums / counts, stored as bf16 hi/lo pair.
// ---------------------------------------------------------------------------
__global__ __launch_bounds__(256) void ge2e_centroids(
    const float* __restrict__ sums, const float* __restrict__ counts,
    unsigned short* __restrict__ ch, unsigned short* __restrict__ cl) {
  int i = blockIdx.x * 256 + threadIdx.x;  // i < C*D
  int row = i >> 8;
  float c = sums[i] / counts[row];
  unsigned short h, l;
  bf16_split(c, h, l);
  ch[i] = h;
  cl[i] = l;
}

// ---------------------------------------------------------------------------
// Kernel 4: fused logits GEMM (bf16 hi/lo WMMA) + log-softmax + NLL.
// Block = 16 rows x 1024 cols. 8 waves, each owns 8 col tiles (128 cols).
// ---------------------------------------------------------------------------
__global__ __launch_bounds__(256) void ge2e_gemm_softmax(
    const float* __restrict__ emb, const int* __restrict__ labels,
    const float* __restrict__ wp, const float* __restrict__ bp,
    const unsigned short* __restrict__ centHi,
    const unsigned short* __restrict__ centLo,
    float* __restrict__ nllAcc) {
  __shared__ float redmax[8][16];
  __shared__ float redsum[8][16];
  __shared__ float rowmax[16];
  __shared__ float labLogit[16];
  __shared__ float nllRow[16];

  const int tid = threadIdx.x;
  const int wid = tid >> 5;
  const int lane = tid & 31;
  const int n = lane & 15;        // col-within-tile / A row (m == n index)
  const int half = lane >> 4;     // lane half selects K sub-chunk / row half
  const int rowBase = blockIdx.x * 16;
  const int colBase = wid * 128;
  const float w = wp[0];
  const float b = bp[0];

  const float* Arow = emb + (size_t)(rowBase + n) * DDIM;

  v8f acc[8];
  v8f z = {};
#pragma unroll
  for (int t = 0; t < 8; ++t) acc[t] = z;

  for (int ks = 0; ks < 8; ++ks) {  // K = 256 in steps of 32
    const int k0 = ks * 32;
    // A fragment (16x32 bf16): lane half picks K chunks [k0+8h..] and
    // [k0+16+8h..]; built in-register from f32 with hi/lo split.
    float4 c00 = *(const float4*)(Arow + k0 + half * 8);
    float4 c01 = *(const float4*)(Arow + k0 + half * 8 + 4);
    float4 c10 = *(const float4*)(Arow + k0 + 16 + half * 8);
    float4 c11 = *(const float4*)(Arow + k0 + 16 + half * 8 + 4);
    AFrag a_hi, a_lo;
    split4(c00, &a_hi.s[0],  &a_lo.s[0]);
    split4(c01, &a_hi.s[4],  &a_lo.s[4]);
    split4(c10, &a_hi.s[8],  &a_lo.s[8]);
    split4(c11, &a_hi.s[12], &a_lo.s[12]);

#pragma unroll
    for (int t = 0; t < 8; ++t) {
      // B fragment (32x16 bf16): lane n = column = centroid row; K chunks
      // [k0+8*half..+8) and [k0+16+8*half..+8).
      const size_t brow = (size_t)(colBase + t * 16 + n) * DDIM;
      const unsigned short* Bh = centHi + brow + k0 + half * 8;
      const unsigned short* Bl = centLo + brow + k0 + half * 8;
      BFrag b_hi, b_lo;
      b_hi.h[0] = *(const v8u16*)(Bh);
      b_hi.h[1] = *(const v8u16*)(Bh + 16);
      b_lo.h[0] = *(const v8u16*)(Bl);
      b_lo.h[1] = *(const v8u16*)(Bl + 16);
      // split-precision: hi*hi + hi*lo + lo*hi
      acc[t] = wmma_bf16(a_hi.v, b_hi.v, acc[t]);
      acc[t] = wmma_bf16(a_hi.v, b_lo.v, acc[t]);
      acc[t] = wmma_bf16(a_lo.v, b_hi.v, acc[t]);
    }
  }

  // logits = w * dot + b (in-register; C/D layout: row = j + 8*half, col = n)
#pragma unroll
  for (int t = 0; t < 8; ++t)
#pragma unroll
    for (int j = 0; j < 8; ++j) acc[t][j] = w * acc[t][j] + b;

  // ---- row max: per-lane over tiles, butterfly over the 16-lane half ----
  float lm[8];
#pragma unroll
  for (int j = 0; j < 8; ++j) {
    float m = -3.402823e38f;
#pragma unroll
    for (int t = 0; t < 8; ++t) m = fmaxf(m, acc[t][j]);
    lm[j] = m;
  }
#pragma unroll
  for (int j = 0; j < 8; ++j) {
    lm[j] = fmaxf(lm[j], __shfl_xor(lm[j], 1, 32));
    lm[j] = fmaxf(lm[j], __shfl_xor(lm[j], 2, 32));
    lm[j] = fmaxf(lm[j], __shfl_xor(lm[j], 4, 32));
    lm[j] = fmaxf(lm[j], __shfl_xor(lm[j], 8, 32));
  }
#pragma unroll
  for (int j = 0; j < 8; ++j)
    if (n == j) redmax[wid][half * 8 + j] = lm[j];
  __syncthreads();
  if (tid < 16) {
    float m = redmax[0][tid];
#pragma unroll
    for (int q = 1; q < 8; ++q) m = fmaxf(m, redmax[q][tid]);
    rowmax[tid] = m;
  }
  __syncthreads();

  // ---- sum of exp + capture label logit (unique owning lane writes) ----
  int lab[8];
#pragma unroll
  for (int j = 0; j < 8; ++j) lab[j] = labels[rowBase + half * 8 + j];

  float ls[8];
#pragma unroll
  for (int j = 0; j < 8; ++j) {
    const float rm = rowmax[half * 8 + j];
    float s = 0.0f;
#pragma unroll
    for (int t = 0; t < 8; ++t) {
      s += __expf(acc[t][j] - rm);
      int col = colBase + t * 16 + n;
      if (col == lab[j]) labLogit[half * 8 + j] = acc[t][j];
    }
    ls[j] = s;
  }
#pragma unroll
  for (int j = 0; j < 8; ++j) {
    ls[j] += __shfl_xor(ls[j], 1, 32);
    ls[j] += __shfl_xor(ls[j], 2, 32);
    ls[j] += __shfl_xor(ls[j], 4, 32);
    ls[j] += __shfl_xor(ls[j], 8, 32);
  }
#pragma unroll
  for (int j = 0; j < 8; ++j)
    if (n == j) redsum[wid][half * 8 + j] = ls[j];
  __syncthreads();
  if (tid < 16) {
    float s = 0.0f;
#pragma unroll
    for (int q = 0; q < 8; ++q) s += redsum[q][tid];
    nllRow[tid] = rowmax[tid] + logf(s) - labLogit[tid];
  }
  __syncthreads();
  if (tid == 0) {
    float s = 0.0f;
#pragma unroll
    for (int r = 0; r < 16; ++r) s += nllRow[r];
    atomicAdd(nllAcc, s);
  }
}

// ---------------------------------------------------------------------------
// Kernel 5: mean
// ---------------------------------------------------------------------------
__global__ void ge2e_finalize(const float* __restrict__ nllAcc,
                              float* __restrict__ out, float invN) {
  if (threadIdx.x == 0 && blockIdx.x == 0) out[0] = nllAcc[0] * invN;
}

// ---------------------------------------------------------------------------
// Host launcher
// ---------------------------------------------------------------------------
extern "C" void kernel_launch(void* const* d_in, const int* in_sizes, int n_in,
                              void* d_out, int out_size, void* d_ws,
                              size_t ws_size, hipStream_t stream) {
  const float* emb = (const float*)d_in[0];
  const int* labels = (const int*)d_in[1];
  const float* wp = (const float*)d_in[2];
  const float* bp = (const float*)d_in[3];
  float* out = (float*)d_out;

  const int N = in_sizes[0] / DDIM;  // 131072

  // Workspace layout (floats):
  //   [0, 262144)          sums   (C*D)
  //   [262144, 263168)     counts (C)
  //   [263168]             nllAcc
  //   from float index 263184 (16B aligned): centHi (C*D u16), centLo
  float* base = (float*)d_ws;
  float* sums = base;
  float* counts = base + (CNUM * DDIM);
  float* nllAcc = base + (CNUM * DDIM) + CNUM;
  unsigned short* centHi = (unsigned short*)(base + (CNUM * DDIM) + CNUM + 16);
  unsigned short* centLo = centHi + (CNUM * DDIM);

  const int nzero = CNUM * DDIM + CNUM + 1;
  ge2e_zero<<<(nzero + 255) / 256, 256, 0, stream>>>(base, nzero);

  ge2e_segsum<<<N, 256, 0, stream>>>(emb, labels, sums, counts);

  ge2e_centroids<<<(CNUM * DDIM) / 256, 256, 0, stream>>>(sums, counts,
                                                          centHi, centLo);

  ge2e_gemm_softmax<<<N / 16, 256, 0, stream>>>(emb, labels, wp, bp, centHi,
                                                centLo, nllAcc);

  ge2e_finalize<<<1, 32, 0, stream>>>(nllAcc, out, 1.0f / (float)N);
}
